// CategoryBranch2_24902220382888
// MI455X (gfx1250) — compile-verified, matrix operand order checked by reading, not run on hardware
//
#include <hip/hip_runtime.h>
#include <hip/hip_bf16.h>
#include <math.h>

typedef __attribute__((ext_vector_type(16))) __bf16 v16bf;
typedef __attribute__((ext_vector_type(8)))  __bf16 v8bf;
typedef __attribute__((ext_vector_type(8)))  float  v8f;

#define BN_EPS 1e-5f

// Build a 16x32 bf16 fragment from two contiguous 8-half (16B) runs:
// halves 0..7  <- K = kc + grp*8 + i        (lo)
// halves 8..15 <- K = kc + 16 + grp*8 + i   (hi)
__device__ __forceinline__ v16bf make_frag(v8bf lo, v8bf hi) {
    v16bf r;
#pragma unroll
    for (int i = 0; i < 8; ++i) { r[i] = lo[i]; r[8 + i] = hi[i]; }
    return r;
}

__device__ __forceinline__ v8bf zero8() {
    v8bf z;
#pragma unroll
    for (int i = 0; i < 8; ++i) z[i] = (__bf16)0.0f;
    return z;
}

__device__ __forceinline__ v8f wmma_bf16(v16bf a, v16bf b, v8f c) {
    return __builtin_amdgcn_wmma_f32_16x16x32_bf16(false, a, false, b, (short)0, c,
                                                   false, false);
}

// ---------------------------------------------------------------------------
// conv1: 1->64, 3x3, pad 1, bias+relu+bn+pool(1,2). Direct fp32 (cheap, ~1.2 GFLOP).
// Output act1: NHWC bf16, idx = ((b*1024+y)*64+xp)*64+co
// ---------------------------------------------------------------------------
__global__ void k_conv1(const float* __restrict__ x, const float* __restrict__ w,
                        const float* __restrict__ bias, const float* __restrict__ g,
                        const float* __restrict__ be, const float* __restrict__ rm,
                        const float* __restrict__ rv, __bf16* __restrict__ act1) {
    int idx = blockIdx.x * blockDim.x + threadIdx.x;
    if (idx >= 8 * 1024 * 64 * 64) return;
    int co = idx & 63;
    int xp = (idx >> 6) & 63;
    int y  = (idx >> 12) & 1023;
    int b  = idx >> 22;
    float scale = g[co] * rsqrtf(rv[co] + BN_EPS);
    float shift = be[co] - rm[co] * scale;
    float best = -INFINITY;
#pragma unroll
    for (int dx = 0; dx < 2; ++dx) {
        int xw = xp * 2 + dx;
        float acc = bias[co];
#pragma unroll
        for (int kh = 0; kh < 3; ++kh)
#pragma unroll
            for (int kw = 0; kw < 3; ++kw) {
                int yy = y + kh - 1, xx = xw + kw - 1;
                if (yy >= 0 && yy < 1024 && xx >= 0 && xx < 128)
                    acc += x[(b * 1024 + yy) * 128 + xx] * w[co * 9 + kh * 3 + kw];
            }
        float v = fmaxf(acc, 0.f) * scale + shift;
        best = fmaxf(best, v);
    }
    act1[idx] = (__bf16)best;
}

// ---------------------------------------------------------------------------
// conv2/conv3 implicit-GEMM WMMA. A = NHWC bf16 activations, K = tap*Cin+ci.
// A 32-wide K chunk never crosses a tap boundary (Cin in {64,128}): one bounds
// check + base per chunk, fragments are two b128 loads. Block stages 32 output
// channels in dynamic LDS ([n][K], K-contiguous); each wave reuses its A
// fragment across 2 N-tiles (2 WMMAs per A load). Epilogue: bias+relu+bn +
// W-pool(1,2) in-register. outMode 0: NHWC; 1: GRU layout (b, t=y, co*16+xp).
// ---------------------------------------------------------------------------
__global__ void k_conv_wmma(const __bf16* __restrict__ actin, const float* __restrict__ w,
                            const float* __restrict__ bias, const float* __restrict__ g,
                            const float* __restrict__ be, const float* __restrict__ rm,
                            const float* __restrict__ rv, __bf16* __restrict__ out,
                            int lgWin, int lgCin, int Cout, int outMode) {
    extern __shared__ __bf16 ldsB[];          // [32][K]
    const int Win = 1 << lgWin;
    const int Cin = 1 << lgCin;
    const int K = 9 << lgCin;
    const int tid = threadIdx.x;
    const int co_base = blockIdx.y * 32;

    // stage weight tile (32 x K) bf16 into LDS; shift/mask index math only
#pragma unroll 1
    for (int n = 0; n < 32; ++n) {
        const float* wn = w + (size_t)(co_base + n) * (Cin * 9);
        for (int k = tid; k < K; k += 256) {
            int tap = k >> lgCin, ci = k & (Cin - 1);
            int kh = tap / 3, kw = tap - kh * 3;     // div by constant 3
            ldsB[n * K + k] = (__bf16)wn[(ci * 3 + kh) * 3 + kw];
        }
    }
    __syncthreads();

    const int lane = tid & 31, wave = tid >> 5;
    const int m_l = lane & 15, grp = lane >> 4, n_l = m_l;
    const int o0 = grp * 8, o1 = 16 + grp * 8;
    const int mtile = blockIdx.x * 8 + wave;
    const int rowA = mtile * 16 + m_l;
    const int lgRow = 10 + lgWin;
    const int bA = rowA >> lgRow;
    const int remA = rowA & ((1 << lgRow) - 1);
    const int yA = remA >> lgWin;
    const int xA = remA & (Win - 1);
    const __bf16* ldsB0 = &ldsB[n_l * K];
    const __bf16* ldsB1 = &ldsB[(16 + n_l) * K];

    v8f acc0 = {}, acc1 = {};
    for (int kc = 0; kc < K; kc += 32) {
        int tap = kc >> lgCin;                 // constant across the 32-chunk
        int cib = kc & (Cin - 1);
        int kh = tap / 3, kw = tap - kh * 3;
        int yy = yA + kh - 1, xx = xA + kw - 1;
        bool valid = (yy >= 0) && (yy < 1024) && (xx >= 0) && (xx < Win);
        v8bf alo, ahi;
        if (valid) {
            const __bf16* base = actin + (((bA * 1024 + yy) * Win + xx) << lgCin) + cib;
            alo = *(const v8bf*)(base + o0);
            ahi = *(const v8bf*)(base + o1);
        } else {
            alo = zero8(); ahi = zero8();
        }
        v16bf a = make_frag(alo, ahi);
        v16bf b0 = make_frag(*(const v8bf*)(ldsB0 + kc + o0),
                             *(const v8bf*)(ldsB0 + kc + o1));
        v16bf b1 = make_frag(*(const v8bf*)(ldsB1 + kc + o0),
                             *(const v8bf*)(ldsB1 + kc + o1));
        acc0 = wmma_bf16(a, b0, acc0);
        acc1 = wmma_bf16(a, b1, acc1);
    }

#pragma unroll
    for (int half = 0; half < 2; ++half) {
        const int co = co_base + half * 16 + n_l;
        const v8f acc = half ? acc1 : acc0;
        float scale = g[co] * rsqrtf(rv[co] + BN_EPS);
        float shift = be[co] - rm[co] * scale;
        float bco = bias[co];
#pragma unroll
        for (int j = 0; j < 8; j += 2) {
            int rowD = mtile * 16 + j + grp * 8;  // even row of the W-pair
            float v0 = fmaxf(acc[j] + bco, 0.f) * scale + shift;
            float v1 = fmaxf(acc[j + 1] + bco, 0.f) * scale + shift;
            float pv = fmaxf(v0, v1);
            int bD = rowD >> lgRow;
            int remD = rowD & ((1 << lgRow) - 1);
            int yD = remD >> lgWin;
            int xp = (remD & (Win - 1)) >> 1;
            if (outMode == 0)
                out[((bD * 1024 + yD) * (Win >> 1) + xp) * Cout + co] = (__bf16)pv;
            else
                out[(bD * 1024 + yD) * (Cout * 16) + co * 16 + xp] = (__bf16)pv;
        }
    }
}

// ---------------------------------------------------------------------------
// C[M,1536] = A[M,K] (bf16) * W[1536,K]^T (bf16, pre-converted). Each wave:
// 1 M-tile x 2 N-tiles, A fragment reused for 2 WMMAs. grid = (M/16, 6).
// ---------------------------------------------------------------------------
__global__ void k_gemm_bf16(const __bf16* __restrict__ A, const __bf16* __restrict__ W,
                            float* __restrict__ C, int K) {
    const int tid = threadIdx.x;
    const int lane = tid & 31, wave = tid >> 5;
    const int m_l = lane & 15, grp = lane >> 4, n_l = m_l;
    const int mtile = blockIdx.x;
    const int nt0 = (blockIdx.y * 8 + wave) * 2;
    const __bf16* Arow  = A + (size_t)(mtile * 16 + m_l) * K;
    const __bf16* Wrow0 = W + (size_t)(nt0 * 16 + n_l) * K;
    const __bf16* Wrow1 = W + (size_t)((nt0 + 1) * 16 + n_l) * K;
    const int o0 = grp * 8, o1 = 16 + grp * 8;
    v8f acc0 = {}, acc1 = {};
    for (int kc = 0; kc < K; kc += 32) {
        if (kc + 256 < K) {
            __builtin_prefetch(Arow + kc + 256, 0, 1);
            __builtin_prefetch(Wrow0 + kc + 256, 0, 1);
            __builtin_prefetch(Wrow1 + kc + 256, 0, 1);
        }
        v16bf a  = make_frag(*(const v8bf*)(Arow + kc + o0),
                             *(const v8bf*)(Arow + kc + o1));
        v16bf b0 = make_frag(*(const v8bf*)(Wrow0 + kc + o0),
                             *(const v8bf*)(Wrow0 + kc + o1));
        v16bf b1 = make_frag(*(const v8bf*)(Wrow1 + kc + o0),
                             *(const v8bf*)(Wrow1 + kc + o1));
        acc0 = wmma_bf16(a, b0, acc0);
        acc1 = wmma_bf16(a, b1, acc1);
    }
#pragma unroll
    for (int j = 0; j < 8; ++j) {
        int row = mtile * 16 + j + grp * 8;
        C[(size_t)row * 1536 + nt0 * 16 + n_l] = acc0[j];
        C[(size_t)row * 1536 + (nt0 + 1) * 16 + n_l] = acc1[j];
    }
}

// ---------------------------------------------------------------------------
// GRU scan: one persistent 1024-thread workgroup per direction.
// h in LDS (fp32 state + bf16 WMMA copy, M padded 8->16). Per step each wave
// owns 3 N-tiles sharing the SAME A rows: A fragment loaded once per K-chunk,
// fed to 3 WMMAs. Then fp32 gate math (GRUCell r,z,n + extra tanh on h).
// ---------------------------------------------------------------------------
__global__ void __launch_bounds__(1024)
k_gru_scan(const float* __restrict__ gx_f, const float* __restrict__ gx_b,
           const __bf16* __restrict__ wh_f, const __bf16* __restrict__ wh_b,
           const float* __restrict__ bi_f, const float* __restrict__ bh_f,
           const float* __restrict__ bi_b, const float* __restrict__ bh_b,
           float* __restrict__ hs_f, float* __restrict__ hs_b, int T) {
    extern __shared__ char smem[];
    float*  h32 = (float*)smem;                                 // 8*512 f32
    float*  gh  = (float*)(smem + 8 * 512 * 4);                 // 8*1536 f32
    __bf16* hbf = (__bf16*)(smem + 8 * 512 * 4 + 8 * 1536 * 4); // 16*512 bf16

    const int rev = blockIdx.x;
    const float*  gx = rev ? gx_b : gx_f;
    const __bf16* wh = rev ? wh_b : wh_f;
    const float*  bi = rev ? bi_b : bi_f;
    const float*  bh = rev ? bh_b : bh_f;
    float* hs = rev ? hs_b : hs_f;

    const int tid = threadIdx.x;
    for (int i = tid; i < 8 * 512; i += 1024) h32[i] = 0.f;
    for (int i = tid; i < 16 * 512; i += 1024) hbf[i] = (__bf16)0.0f;
    __syncthreads();

    const int lane = tid & 31, wave = tid >> 5;
    const int m_l = lane & 15, grp = lane >> 4, n_l = m_l;
    const int o0 = grp * 8, o1 = 16 + grp * 8;
    const __bf16* hrow = hbf + m_l * 512;
    const __bf16* w0 = wh + (size_t)((wave      ) * 16 + n_l) * 512;
    const __bf16* w1 = wh + (size_t)((wave + 32 ) * 16 + n_l) * 512;
    const __bf16* w2 = wh + (size_t)((wave + 64 ) * 16 + n_l) * 512;

    for (int t = 0; t < T; ++t) {
        // gh[8x1536] = h[8x512 (pad 16)] * wh^T : 3 tiles/wave sharing A
        v8f acc0 = {}, acc1 = {}, acc2 = {};
#pragma unroll 4
        for (int kc = 0; kc < 512; kc += 32) {
            v16bf a = make_frag(*(const v8bf*)(hrow + kc + o0),
                                *(const v8bf*)(hrow + kc + o1));
            v16bf b0 = make_frag(*(const v8bf*)(w0 + kc + o0),
                                 *(const v8bf*)(w0 + kc + o1));
            v16bf b1 = make_frag(*(const v8bf*)(w1 + kc + o0),
                                 *(const v8bf*)(w1 + kc + o1));
            v16bf b2 = make_frag(*(const v8bf*)(w2 + kc + o0),
                                 *(const v8bf*)(w2 + kc + o1));
            acc0 = wmma_bf16(a, b0, acc0);
            acc1 = wmma_bf16(a, b1, acc1);
            acc2 = wmma_bf16(a, b2, acc2);
        }
#pragma unroll
        for (int j = 0; j < 8; ++j) {
            int m = j + grp * 8;
            if (m < 8) {
                gh[m * 1536 + (wave     ) * 16 + n_l] = acc0[j];
                gh[m * 1536 + (wave + 32) * 16 + n_l] = acc1[j];
                gh[m * 1536 + (wave + 64) * 16 + n_l] = acc2[j];
            }
        }
        __syncthreads();

        int t_nat = rev ? (T - 1 - t) : t;
#pragma unroll
        for (int e = 0; e < 4; ++e) {
            int idx = tid + e * 1024;   // (b, g)
            int bB = idx >> 9;
            int gg = idx & 511;
            size_t gxrow = ((size_t)bB * T + t_nat) * 1536;
            float ghr = gh[bB * 1536 + gg] + bh[gg];
            float ghz = gh[bB * 1536 + 512 + gg] + bh[512 + gg];
            float ghn = gh[bB * 1536 + 1024 + gg] + bh[1024 + gg];
            float gxr = gx[gxrow + gg] + bi[gg];
            float gxz = gx[gxrow + 512 + gg] + bi[512 + gg];
            float gxn = gx[gxrow + 1024 + gg] + bi[1024 + gg];
            float r = 1.f / (1.f + __expf(-(gxr + ghr)));
            float z = 1.f / (1.f + __expf(-(gxz + ghz)));
            float n = tanhf(gxn + r * ghn);
            float hp = h32[bB * 512 + gg];
            float hn = tanhf((1.f - z) * n + z * hp);
            h32[bB * 512 + gg] = hn;
            hbf[bB * 512 + gg] = (__bf16)hn;
            hs[((size_t)bB * T + t) * 512 + gg] = hn;
        }
        __syncthreads();
    }
}

// ---------------------------------------------------------------------------
// helpers: f32->bf16 weight conversion, sum + time-subsample (stride 2)
// ---------------------------------------------------------------------------
__global__ void k_f32_to_bf16(const float* __restrict__ in, __bf16* __restrict__ out, int n) {
    int i = blockIdx.x * blockDim.x + threadIdx.x;
    if (i < n) out[i] = (__bf16)in[i];
}

__global__ void k_sumsub_bf(const float* __restrict__ hf, const float* __restrict__ hb,
                            __bf16* __restrict__ out, int Tin, int n) {
    int i = blockIdx.x * blockDim.x + threadIdx.x;
    if (i >= n) return;
    int h = i & 511;
    int s = (i >> 9) % (Tin >> 1);
    int b = i / ((Tin >> 1) * 512);
    size_t src = ((size_t)b * Tin + 2 * s) * 512 + h;
    out[i] = (__bf16)(hf[src] + hb[src]);
}

__global__ void k_sumsub_f32(const float* __restrict__ hf, const float* __restrict__ hb,
                             float* __restrict__ out, int Tin, int n) {
    int i = blockIdx.x * blockDim.x + threadIdx.x;
    if (i >= n) return;
    int h = i & 511;
    int s = (i >> 9) % (Tin >> 1);
    int b = i / ((Tin >> 1) * 512);
    size_t src = ((size_t)b * Tin + 2 * s) * 512 + h;
    out[i] = hf[src] + hb[src];
}

// ---------------------------------------------------------------------------
extern "C" void kernel_launch(void* const* d_in, const int* in_sizes, int n_in,
                              void* d_out, int out_size, void* d_ws, size_t ws_size,
                              hipStream_t stream) {
    (void)in_sizes; (void)n_in; (void)out_size; (void)ws_size;
    const float* x    = (const float*)d_in[0];
    const float* cw1  = (const float*)d_in[1];
    const float* cb1  = (const float*)d_in[2];
    const float* g1   = (const float*)d_in[3];
    const float* be1  = (const float*)d_in[4];
    const float* rm1  = (const float*)d_in[5];
    const float* rv1  = (const float*)d_in[6];
    const float* cw2  = (const float*)d_in[7];
    const float* cb2  = (const float*)d_in[8];
    const float* g2   = (const float*)d_in[9];
    const float* be2  = (const float*)d_in[10];
    const float* rm2  = (const float*)d_in[11];
    const float* rv2  = (const float*)d_in[12];
    const float* cw3  = (const float*)d_in[13];
    const float* cb3  = (const float*)d_in[14];
    const float* g3   = (const float*)d_in[15];
    const float* be3  = (const float*)d_in[16];
    const float* rm3  = (const float*)d_in[17];
    const float* rv3  = (const float*)d_in[18];
    const float* wif1 = (const float*)d_in[19];
    const float* whf1 = (const float*)d_in[20];
    const float* bif1 = (const float*)d_in[21];
    const float* bhf1 = (const float*)d_in[22];
    const float* wib1 = (const float*)d_in[23];
    const float* whb1 = (const float*)d_in[24];
    const float* bib1 = (const float*)d_in[25];
    const float* bhb1 = (const float*)d_in[26];
    const float* wif2 = (const float*)d_in[27];
    const float* whf2 = (const float*)d_in[28];
    const float* bif2 = (const float*)d_in[29];
    const float* bhf2 = (const float*)d_in[30];
    const float* wib2 = (const float*)d_in[31];
    const float* whb2 = (const float*)d_in[32];
    const float* bib2 = (const float*)d_in[33];
    const float* bhb2 = (const float*)d_in[34];

    char* ws = (char*)d_ws;
    // regions A/B/C (64MB each) with lifetime-based reuse + D for bf16 weights
    const size_t OFF_B = 67108864, OFF_C = 134217728, OFF_D = 201326592;
    __bf16* act1  = (__bf16*)(ws + 0);                 // conv1 out (dead after conv2)
    __bf16* act2  = (__bf16*)(ws + OFF_B);             // conv2 out (dead after conv3)
    __bf16* gruin = (__bf16*)(ws + OFF_C);             // conv3 out, GRU layout (dead after gx1)
    float* gx1f = (float*)(ws + 0);                    // reuses A
    float* gx1b = (float*)(ws + OFF_B);                // reuses B
    float* hs1f = (float*)(ws + OFF_C);                // reuses C (gruin dead)
    float* hs1b = (float*)(ws + OFF_C + 16777216);
    __bf16* x2  = (__bf16*)(ws + OFF_C + 33554432);
    float* hs2f = (float*)(ws + OFF_C + 37748736);
    float* hs2b = (float*)(ws + OFF_C + 46137344);
    float* gx2f = (float*)(ws + 0);                    // reuses A (gx1 dead after scan1)
    float* gx2b = (float*)(ws + OFF_B);
    __bf16* whbf_f1 = (__bf16*)(ws + OFF_D);
    __bf16* whbf_b1 = (__bf16*)(ws + OFF_D + 1572864);
    __bf16* whbf_f2 = (__bf16*)(ws + OFF_D + 3145728);
    __bf16* whbf_b2 = (__bf16*)(ws + OFF_D + 4718592);
    __bf16* wibf_f1 = (__bf16*)(ws + OFF_D + 6291456);           // 12.58MB
    __bf16* wibf_b1 = (__bf16*)(ws + OFF_D + 6291456 + 12582912);
    __bf16* wibf_f2 = (__bf16*)(ws + OFF_D + 6291456 + 25165824);
    __bf16* wibf_b2 = (__bf16*)(ws + OFF_D + 6291456 + 26738688);

    // weights to bf16 (recurrent 1536x512, input 1536x4096 / 1536x512)
    k_f32_to_bf16<<<3072, 256, 0, stream>>>(whf1, whbf_f1, 786432);
    k_f32_to_bf16<<<3072, 256, 0, stream>>>(whb1, whbf_b1, 786432);
    k_f32_to_bf16<<<3072, 256, 0, stream>>>(whf2, whbf_f2, 786432);
    k_f32_to_bf16<<<3072, 256, 0, stream>>>(whb2, whbf_b2, 786432);
    k_f32_to_bf16<<<24576, 256, 0, stream>>>(wif1, wibf_f1, 6291456);
    k_f32_to_bf16<<<24576, 256, 0, stream>>>(wib1, wibf_b1, 6291456);
    k_f32_to_bf16<<<3072, 256, 0, stream>>>(wif2, wibf_f2, 786432);
    k_f32_to_bf16<<<3072, 256, 0, stream>>>(wib2, wibf_b2, 786432);

    // conv stack (dynamic LDS = 32 channels * K * 2B)
    k_conv1<<<131072, 256, 0, stream>>>(x, cw1, cb1, g1, be1, rm1, rv1, act1);
    k_conv_wmma<<<dim3(4096, 4), 256, 36864, stream>>>(act1, cw2, cb2, g2, be2, rm2, rv2,
                                                       act2, /*lgWin=*/6, /*lgCin=*/6,
                                                       /*Cout=*/128, /*outMode=*/0);
    k_conv_wmma<<<dim3(2048, 8), 256, 73728, stream>>>(act2, cw3, cb3, g3, be3, rm3, rv3,
                                                       gruin, /*lgWin=*/5, /*lgCin=*/7,
                                                       /*Cout=*/256, /*outMode=*/1);

    // BiGRU layer 1: gx = X @ Wi^T (M=8192, K=4096), then scan T=1024
    k_gemm_bf16<<<dim3(512, 6), 256, 0, stream>>>(gruin, wibf_f1, gx1f, 4096);
    k_gemm_bf16<<<dim3(512, 6), 256, 0, stream>>>(gruin, wibf_b1, gx1b, 4096);
    k_gru_scan<<<2, 1024, 81920, stream>>>(gx1f, gx1b, whbf_f1, whbf_b1,
                                           bif1, bhf1, bib1, bhb1, hs1f, hs1b, 1024);
    k_sumsub_bf<<<8192, 256, 0, stream>>>(hs1f, hs1b, x2, 1024, 2097152);

    // BiGRU layer 2: M=4096, K=512, scan T=512
    k_gemm_bf16<<<dim3(256, 6), 256, 0, stream>>>(x2, wibf_f2, gx2f, 512);
    k_gemm_bf16<<<dim3(256, 6), 256, 0, stream>>>(x2, wibf_b2, gx2b, 512);
    k_gru_scan<<<2, 1024, 81920, stream>>>(gx2f, gx2b, whbf_f2, whbf_b2,
                                           bif2, bhf2, bib2, bhb2, hs2f, hs2b, 512);
    k_sumsub_f32<<<4096, 256, 0, stream>>>(hs2f, hs2b, (float*)d_out, 512, 1048576);
}
